// MHA_8306466750773
// MI455X (gfx1250) — matrix operand order repaired and checked by reading, not compiled, not run on hardware
//
#include <hip/hip_runtime.h>

// ---------------------------------------------------------------------------
// MHA: qkv projection + 2-head attention, f16 WMMA (v_wmma_f32_16x16x32_f16)
// + gfx1250 async global->LDS copies (GLOBAL_LOAD_ASYNC_TO_LDS_B128, ASYNCcnt)
// B=16, C=128, P=512, T=32, H=2, D=64, E=D*T=2048, SCALE=1/8
// ---------------------------------------------------------------------------

typedef __attribute__((ext_vector_type(16))) _Float16 v16h;
typedef __attribute__((ext_vector_type(8)))  _Float16 v8h;
typedef __attribute__((ext_vector_type(8)))  float    v8f;
typedef __attribute__((ext_vector_type(4)))  int      v4i;

#define B_   16
#define C_   128
#define P_   512
#define T_   32
#define H_   2
#define D_   64
#define E_   2048      // D_*T_
#define O3_  384       // 3*C_
#define BH_  32        // B_*H_
#define SCALE_ 0.125f  // D_^-0.5

// ---- async global->LDS copy (16 bytes per lane) ---------------------------
#if defined(__gfx1250__) && __has_builtin(__builtin_amdgcn_global_load_async_to_lds_b128)
#define ASYNC_BUILTIN 1
// builtin param 0 is ptr-to-int4 in addrspace(1); param 1 LDS side
typedef __attribute__((address_space(1))) v4i gv4i;
typedef __attribute__((address_space(3))) v4i lv4i;
#endif

__device__ __forceinline__ void async_cp16(const _Float16* g, _Float16* l) {
#if defined(ASYNC_BUILTIN)
    __builtin_amdgcn_global_load_async_to_lds_b128(
        (gv4i*)g, (lv4i*)l, 0, 0);
#elif defined(__gfx1250__)
    unsigned lds_off = (unsigned)(size_t)(__attribute__((address_space(3))) void*)l;
    asm volatile("global_load_async_to_lds_b128 %0, %1, off"
                 :: "v"(lds_off), "v"((unsigned long long)(size_t)g)
                 : "memory");
#else
    *(v8h*)l = *(const v8h*)g;   // host pass / other targets
#endif
}

__device__ __forceinline__ void wait_async() {
#if defined(__gfx1250__)
#if __has_builtin(__builtin_amdgcn_s_wait_asynccnt)
    __builtin_amdgcn_s_wait_asynccnt(0);
#else
    asm volatile("s_wait_asynccnt 0x0" ::: "memory");
#endif
#endif
}

// ---- fragment helpers (per ISA 7.12.2 layouts, wave32) --------------------
// A 16x32 f16: lane<16 -> M=lane, K in {kb..kb+7, kb+16..kb+23}, kb = (lane&16)?8:0
// B 32x16 f16: lane<16 -> N=lane, K in {kb..kb+7, kb+16..kb+23}, kb = (lane&16)?8:0
// C/D 16x16 f32: elem i -> M = i + 8*(lane>=16), N = lane&15

__device__ __forceinline__ v16h load_a_frag(const _Float16* row, int k0, int lane) {
    const int kb = (lane & 16) ? 8 : 0;
    v8h lo = *(const v8h*)(row + k0 + kb);
    v8h hi = *(const v8h*)(row + k0 + kb + 16);
    return __builtin_shufflevector(lo, hi, 0,1,2,3,4,5,6,7,8,9,10,11,12,13,14,15);
}

__device__ __forceinline__ v16h load_b_frag(const _Float16* row, int k0, int lane) {
    const int kb = (lane & 16) ? 8 : 0;
    v8h lo = *(const v8h*)(row + k0 + kb);
    v8h hi = *(const v8h*)(row + k0 + kb + 16);
    return __builtin_shufflevector(lo, hi, 0,1,2,3,4,5,6,7,8,9,10,11,12,13,14,15);
}

// ---------------------------------------------------------------------------
// Kernel 1: qkv = x^T W^T + b   (M tile = one (b,p) pair: 32 t-rows, K=C=128)
// N tile = 192 (grid.y = 2).  Writes q/k/v as f16 in (b,h,p, e=d*32+t) layout.
// (keeps the VALU staging path: f32 -> f16 conversion happens during the copy)
// ---------------------------------------------------------------------------
__global__ __launch_bounds__(256) void qkv_proj(
    const float* __restrict__ x, const float* __restrict__ W,
    const float* __restrict__ bias,
    _Float16* __restrict__ q, _Float16* __restrict__ k, _Float16* __restrict__ v)
{
    __shared__ alignas(32) _Float16 aS[32][128];    // [m=t][k=c]
    __shared__ alignas(32) _Float16 bS[192][128];   // [n][k=c]

    const int bp  = blockIdx.x;            // b*P + p
    const int b   = bp >> 9;
    const int p   = bp & 511;
    const int n0  = blockIdx.y * 192;
    const int tid = threadIdx.x;

    const float* xb = x + (((size_t)b * C_ * P_) + p) * T_;   // x[b,0,p,0]

    // stage x tile (128 c x 32 t), f32 -> f16, as aS[t][c]
    #pragma unroll
    for (int i = 0; i < 4; ++i) {
        int lin = (tid + 256 * i) * 4;                 // 4096 elems, float4
        int c = lin >> 5, t0 = lin & 31;
        float4 x4 = *(const float4*)&xb[(size_t)c * (P_ * T_) + t0];
        aS[t0 + 0][c] = (_Float16)x4.x;
        aS[t0 + 1][c] = (_Float16)x4.y;
        aS[t0 + 2][c] = (_Float16)x4.z;
        aS[t0 + 3][c] = (_Float16)x4.w;
    }
    // stage W tile (192 n x 128 c) as bS[n][c] (W row-major in c: direct copy)
    #pragma unroll
    for (int i = 0; i < 24; ++i) {
        int lin = (tid + 256 * i) * 4;                 // 24576 elems
        int n = lin >> 7, c0 = lin & 127;
        float4 w4 = *(const float4*)&W[(size_t)(n0 + n) * C_ + c0];
        bS[n][c0 + 0] = (_Float16)w4.x;
        bS[n][c0 + 1] = (_Float16)w4.y;
        bS[n][c0 + 2] = (_Float16)w4.z;
        bS[n][c0 + 3] = (_Float16)w4.w;
    }
    __syncthreads();

    const int wave = tid >> 5, lane = tid & 31;
    const int tm = wave >> 2;               // 0..1 -> m-subtile (16 t-rows)
    const int mrow = tm * 16 + (lane & 15);

    v8f acc[3] = { {0,0,0,0,0,0,0,0}, {0,0,0,0,0,0,0,0}, {0,0,0,0,0,0,0,0} };

    #pragma unroll
    for (int kk = 0; kk < 4; ++kk) {        // K = 128 in 4 steps of 32
        const int k0 = kk * 32;
        v16h af = load_a_frag(&aS[mrow][0], k0, lane);
        #pragma unroll
        for (int r = 0; r < 3; ++r) {
            int ncol = ((wave & 3) + 4 * r) * 16 + (lane & 15);
            v16h bf = load_b_frag(&bS[ncol][0], k0, lane);
            acc[r] = __builtin_amdgcn_wmma_f32_16x16x32_f16(
                false, af, false, bf, (short)0, acc[r], false, false);
        }
    }

    // epilogue: +bias, scatter into q/k/v f16 (b,h,p, e = d*T + t)
    #pragma unroll
    for (int r = 0; r < 3; ++r) {
        int ncol = ((wave & 3) + 4 * r) * 16 + (lane & 15);
        int o = n0 + ncol;                          // 0..383
        float bb = bias[o];
        int which = o >> 7;                         // 0=q 1=k 2=v
        int h     = (o >> 6) & 1;
        int d     = o & 63;
        _Float16* dst = (which == 0) ? q : (which == 1 ? k : v);
        size_t rowbase = (((size_t)(b * H_ + h) * P_) + p) * E_ + (size_t)d * T_;
        int mb = tm * 16 + ((lane >> 4) << 3);
        #pragma unroll
        for (int i = 0; i < 8; ++i) {
            int t = mb + i;                         // t == local m row
            dst[rowbase + t] = (_Float16)(acc[r][i] + bb);
        }
    }
}

// ---------------------------------------------------------------------------
// Kernel 2: dots = SCALE * q @ k^T  per (b,h).  M=N=512, K=E=2048.
// Block tile 64x64, K chunked by 64, 2 m-subtiles per wave.
// q/k tiles staged with async global->LDS copies (no VGPR round-trip).
// ---------------------------------------------------------------------------
__global__ __launch_bounds__(256) void qk_dots(
    const _Float16* __restrict__ q, const _Float16* __restrict__ k,
    float* __restrict__ dots)
{
    __shared__ alignas(32) _Float16 aS[64][64];   // q  [m][e]
    __shared__ alignas(32) _Float16 bS[64][64];   // k  [n][e]

    const int m0 = blockIdx.x * 64;
    const int n0 = blockIdx.y * 64;
    const int bh = blockIdx.z;
    const int tid = threadIdx.x;
    const int wave = tid >> 5, lane = tid & 31;
    const int sm0 = (wave >> 2) * 32;             // wave owns rows sm0..sm0+31
    const int sn  = (wave & 3) * 16;
    const int ncol = sn + (lane & 15);

    const _Float16* qb = q + (size_t)(bh * P_ + m0) * E_;
    const _Float16* kb = k + (size_t)(bh * P_ + n0) * E_;

    v8f acc[2] = { {0,0,0,0,0,0,0,0}, {0,0,0,0,0,0,0,0} };

    for (int e0 = 0; e0 < E_; e0 += 64) {
        __syncthreads();
        #pragma unroll
        for (int i = 0; i < 2; ++i) {             // 4096 halves each, 16B/lane
            int idx = tid + 256 * i;
            int r = idx >> 3, c0 = (idx & 7) * 8;
            async_cp16(&qb[(size_t)r * E_ + e0 + c0], &aS[r][c0]);
            async_cp16(&kb[(size_t)r * E_ + e0 + c0], &bS[r][c0]);
        }
        if (e0 + 64 < E_)
            __builtin_prefetch(&qb[(size_t)(tid >> 3) * E_ + e0 + 64], 0, 1);
        wait_async();
        __syncthreads();

        #pragma unroll
        for (int kk = 0; kk < 2; ++kk) {          // chunk = 2 WMMA K-steps
            const int k0 = kk * 32;
            v16h bf = load_b_frag(&bS[ncol][0], k0, lane);
            #pragma unroll
            for (int s = 0; s < 2; ++s) {
                int mrow = sm0 + s * 16 + (lane & 15);
                v16h af = load_a_frag(&aS[mrow][0], k0, lane);
                acc[s] = __builtin_amdgcn_wmma_f32_16x16x32_f16(
                    false, af, false, bf, (short)0, acc[s], false, false);
            }
        }
    }

    #pragma unroll
    for (int s = 0; s < 2; ++s) {
        int mb = sm0 + s * 16 + ((lane >> 4) << 3);
        #pragma unroll
        for (int i = 0; i < 8; ++i) {
            int m = m0 + mb + i;
            dots[((size_t)bh * P_ + m) * P_ + n0 + ncol] = acc[s][i] * SCALE_;
        }
    }
}

// ---------------------------------------------------------------------------
// Kernel 3: row softmax (512 cols, f32 in -> f16 out), one block per row
// ---------------------------------------------------------------------------
__global__ __launch_bounds__(256) void softmax_rows(
    const float* __restrict__ dots, _Float16* __restrict__ attn)
{
    __shared__ float red[256];
    const size_t row = (size_t)blockIdx.y * P_ + blockIdx.x;  // bh*512 + p
    const int tid = threadIdx.x;
    const float* rp = dots + row * P_;

    float x0 = rp[tid], x1 = rp[tid + 256];
    red[tid] = fmaxf(x0, x1);
    __syncthreads();
    for (int s = 128; s > 0; s >>= 1) {
        if (tid < s) red[tid] = fmaxf(red[tid], red[tid + s]);
        __syncthreads();
    }
    const float mx = red[0];
    __syncthreads();

    float e0 = __expf(x0 - mx), e1 = __expf(x1 - mx);
    red[tid] = e0 + e1;
    __syncthreads();
    for (int s = 128; s > 0; s >>= 1) {
        if (tid < s) red[tid] += red[tid + s];
        __syncthreads();
    }
    const float inv = 1.0f / red[0];

    _Float16* ap = attn + row * P_;
    ap[tid]       = (_Float16)(e0 * inv);
    ap[tid + 256] = (_Float16)(e1 * inv);
}

// ---------------------------------------------------------------------------
// Kernel 4: out = attn @ v  per (b,h). M=512(p), N=2048(e), K=512.
// attn tile staged async; v tile transposed manually into LDS.
// Scatter f32 result into out[b, c=h*64+d, p, t] with e = d*32 + t.
// ---------------------------------------------------------------------------
__global__ __launch_bounds__(256) void av_out(
    const _Float16* __restrict__ attn, const _Float16* __restrict__ v,
    float* __restrict__ out)
{
    __shared__ alignas(32) _Float16 aS[64][64];   // attn [m][kp]
    __shared__ alignas(32) _Float16 bS[64][64];   // v^T  [n=e][kp]

    const int m0 = blockIdx.x * 64;
    const int n0 = blockIdx.y * 64;               // e tile
    const int bh = blockIdx.z;
    const int b  = bh >> 1, h = bh & 1;
    const int tid = threadIdx.x;
    const int wave = tid >> 5, lane = tid & 31;
    const int sm0 = (wave >> 2) * 32;
    const int sn  = (wave & 3) * 16;
    const int ncol = sn + (lane & 15);

    const _Float16* ab = attn + (size_t)(bh * P_ + m0) * P_;
    const _Float16* vb = v    + (size_t)bh * P_ * E_;

    v8f acc[2] = { {0,0,0,0,0,0,0,0}, {0,0,0,0,0,0,0,0} };

    for (int kp0 = 0; kp0 < P_; kp0 += 64) {
        __syncthreads();
        #pragma unroll
        for (int i = 0; i < 2; ++i) {
            int idx = tid + 256 * i;
            int r = idx >> 3, c0 = (idx & 7) * 8;
            // attn tile: straight row copy -> async to LDS
            async_cp16(&ab[(size_t)r * P_ + kp0 + c0], &aS[r][c0]);
            // v tile: transpose into bS[e][kp] (coalesced global read in e)
            int kl = idx >> 3, nl0 = (idx & 7) * 8;
            v8h t = *(const v8h*)&vb[(size_t)(kp0 + kl) * E_ + n0 + nl0];
            #pragma unroll
            for (int j = 0; j < 8; ++j) bS[nl0 + j][kl] = t[j];
        }
        wait_async();
        __syncthreads();

        #pragma unroll
        for (int kk = 0; kk < 2; ++kk) {
            const int k0 = kk * 32;
            v16h bf = load_b_frag(&bS[ncol][0], k0, lane);
            #pragma unroll
            for (int s = 0; s < 2; ++s) {
                int mrow = sm0 + s * 16 + (lane & 15);
                v16h af = load_a_frag(&aS[mrow][0], k0, lane);
                acc[s] = __builtin_amdgcn_wmma_f32_16x16x32_f16(
                    false, af, false, bf, (short)0, acc[s], false, false);
            }
        }
    }

    const int e = n0 + ncol;
    const int d = e >> 5, t = e & 31;
    const int c = h * D_ + d;
    #pragma unroll
    for (int s = 0; s < 2; ++s) {
        int mb = sm0 + s * 16 + ((lane >> 4) << 3);
        #pragma unroll
        for (int i = 0; i < 8; ++i) {
            int p = m0 + mb + i;
            out[(((size_t)(b * C_ + c)) * P_ + p) * T_ + t] = acc[s][i];
        }
    }
}

// ---------------------------------------------------------------------------
// launch
// ---------------------------------------------------------------------------
extern "C" void kernel_launch(void* const* d_in, const int* in_sizes, int n_in,
                              void* d_out, int out_size, void* d_ws, size_t ws_size,
                              hipStream_t stream) {
    const float* x    = (const float*)d_in[0];
    const float* W    = (const float*)d_in[1];
    const float* bias = (const float*)d_in[2];
    float* out = (float*)d_out;

    // workspace carve: q,k,v (f16, 64MB each) | dots (f32, 32MB) | attn (f16, 16MB)
    const size_t QKV  = (size_t)B_ * H_ * P_ * E_;      // 33,554,432
    const size_t DOTS = (size_t)BH_ * P_ * P_;          //  8,388,608
    char* ws = (char*)d_ws;
    _Float16* q    = (_Float16*)ws;
    _Float16* k    = q + QKV;
    _Float16* v    = k + QKV;
    float*    dots = (float*)(ws + 3 * QKV * sizeof(_Float16));
    _Float16* attn = (_Float16*)((char*)dots + DOTS * sizeof(float));

    qkv_proj    <<<dim3(B_ * P_, 2),  256, 0, stream>>>(x, W, bias, q, k, v);
    qk_dots     <<<dim3(8, 8, BH_),   256, 0, stream>>>(q, k, dots);
    softmax_rows<<<dim3(P_, BH_),     256, 0, stream>>>(dots, attn);
    av_out      <<<dim3(8, 32, BH_),  256, 0, stream>>>(attn, v, out);

    (void)in_sizes; (void)n_in; (void)out_size; (void)ws_size;
}
